// ContrastiveLoss_29643864277143
// MI455X (gfx1250) — matrix operand order repaired and checked by reading, not compiled, not run on hardware
//
#include <hip/hip_runtime.h>

typedef __attribute__((ext_vector_type(16))) _Float16 v16h;
typedef __attribute__((ext_vector_type(8)))  float    v8f;
typedef __attribute__((ext_vector_type(4)))  float    v4f;   // clang vector -> OK for nontemporal builtin

#define DFIX 1024
#define KCN  (DFIX / 32)     // 32 K-chunks of 32
#define SMAX 160             // padded sample slots (S=129 -> 9 tiles of 16)

__device__ __forceinline__ float sq4(v4f v) {
    return v.x * v.x + v.y * v.y + v.z * v.z + v.w * v.w;
}

__device__ __forceinline__ void cvt4(v16h& a, int o, v4f v) {
    a[o + 0] = (_Float16)v.x; a[o + 1] = (_Float16)v.y;
    a[o + 2] = (_Float16)v.z; a[o + 3] = (_Float16)v.w;
}

__global__ void __launch_bounds__(512)
contrastive_wmma_kernel(const float* __restrict__ anchor,
                        const float* __restrict__ positives,
                        const float* __restrict__ negatives,
                        float* __restrict__ per_batch,
                        int P, int S)
{
    const int b    = blockIdx.x;
    const int tid  = threadIdx.x;
    const int lane = tid & 31;
    const int wave = tid >> 5;
    const int hi   = lane >> 4;     // which K-half this lane owns
    const int m    = lane & 15;     // tile row (sample within tile)

    __shared__ _Float16 ah[DFIX];   // anchor, f16
    __shared__ float    red[512];   // block reduction scratch
    __shared__ float    dotv[SMAX]; // per-sample dot(a, s)
    __shared__ float    ns2v[SMAX]; // per-sample ||s||^2
    __shared__ float    na2_sh;     // ||a||^2

    // ---- Stage anchor into LDS as f16; accumulate ||a||^2 partials ----
    const float* arow = anchor + (size_t)b * DFIX;
    float na2p = 0.f;
    for (int i4 = tid; i4 < DFIX / 4; i4 += blockDim.x) {
        v4f v = *(const v4f*)(arow + i4 * 4);
        ah[i4 * 4 + 0] = (_Float16)v.x;
        ah[i4 * 4 + 1] = (_Float16)v.y;
        ah[i4 * 4 + 2] = (_Float16)v.z;
        ah[i4 * 4 + 3] = (_Float16)v.w;
        na2p += sq4(v);
    }
    red[tid] = na2p;
    __syncthreads();
    if (wave == 0) {
        float s = 0.f;
        for (int i = lane; i < (int)blockDim.x; i += 32) s += red[i];
        for (int off = 16; off > 0; off >>= 1) s += __shfl_xor(s, off, 32);
        if (lane == 0) na2_sh = s;
    }

    // ---- WMMA: 16 samples (M) x anchor broadcast (N), K = 1024 in 32-chunks ----
    // Sample s: 0 -> last positive row, 1..S-1 -> negatives row s-1.
    const int s_idx = wave * 16 + m;
    const int s_eff = (s_idx < S) ? s_idx : (S - 1);   // clamp; results unused
    const float* row;
    if (s_eff == 0) row = positives + ((size_t)b * P + (P - 1)) * DFIX;
    else            row = negatives + ((size_t)b * (S - 1) + (s_eff - 1)) * DFIX;

    v8f c = {};                                   // f32 accumulator (16x16)
    float ns0 = 0.f, ns1 = 0.f, ns2 = 0.f, ns3 = 0.f;  // independent ||s||^2 partials

    // Software pipeline: double-buffer A (global, non-temporal) and B (LDS).
    // A-layout K groups for this lane: [hi*8, hi*8+8) and [16+hi*8, 16+hi*8+8).
    const float* base0 = row + hi * 8;
    v4f a0 = __builtin_nontemporal_load((const v4f*)(base0 + 0));
    v4f a1 = __builtin_nontemporal_load((const v4f*)(base0 + 4));
    v4f a2 = __builtin_nontemporal_load((const v4f*)(base0 + 16));
    v4f a3 = __builtin_nontemporal_load((const v4f*)(base0 + 20));
    v16h bb;   // B-layout: lane holds anchor K = kc*32 + hi*8 + (j&7) + (j>>3)*16 (bcast over cols)
    {
        const _Float16* bp = &ah[hi * 8];
        #pragma unroll
        for (int j = 0; j < 16; ++j) bb[j] = bp[(j & 7) + ((j >> 3) << 4)];
    }

    #pragma unroll 4
    for (int kc = 0; kc < KCN - 1; ++kc) {
        v4f g0 = a0, g1 = a1, g2 = a2, g3 = a3;
        v16h bcur = bb;

        // Issue next chunk's loads before consuming this chunk.
        const float* nb = row + (kc + 1) * 32 + hi * 8;
        a0 = __builtin_nontemporal_load((const v4f*)(nb + 0));
        a1 = __builtin_nontemporal_load((const v4f*)(nb + 4));
        a2 = __builtin_nontemporal_load((const v4f*)(nb + 16));
        a3 = __builtin_nontemporal_load((const v4f*)(nb + 20));
        {
            const _Float16* bp = &ah[(kc + 1) * 32 + hi * 8];
            #pragma unroll
            for (int j = 0; j < 16; ++j) bb[j] = bp[(j & 7) + ((j >> 3) << 4)];
        }

        ns0 += sq4(g0); ns1 += sq4(g1); ns2 += sq4(g2); ns3 += sq4(g3);

        v16h a;
        cvt4(a, 0, g0); cvt4(a, 4, g1); cvt4(a, 8, g2); cvt4(a, 12, g3);

        c = __builtin_amdgcn_wmma_f32_16x16x32_f16(
                false, a, false, bcur, (short)0, c, false, false);
    }
    {   // peeled final chunk (kc = KCN-1)
        ns0 += sq4(a0); ns1 += sq4(a1); ns2 += sq4(a2); ns3 += sq4(a3);
        v16h a;
        cvt4(a, 0, a0); cvt4(a, 4, a1); cvt4(a, 8, a2); cvt4(a, 12, a3);
        c = __builtin_amdgcn_wmma_f32_16x16x32_f16(
                false, a, false, bb, (short)0, c, false, false);
    }

    // ---- Fold ||s||^2 across the two K-half lanes; spill dots from N=0 lanes ----
    float ns2p = (ns0 + ns1) + (ns2 + ns3);
    ns2p += __shfl_xor(ns2p, 16, 32);
    if (lane < 16) ns2v[wave * 16 + lane] = ns2p;
    if (lane == 0 || lane == 16) {
        const int mbase = wave * 16 + hi * 8;  // C rows: lane0 -> M=0..7, lane16 -> M=8..15
        #pragma unroll
        for (int r = 0; r < 8; ++r) dotv[mbase + r] = c[r];
    }
    __syncthreads();

    // ---- Softmax over S sims by wave 0; per-batch loss out ----
    if (wave == 0) {
        const float na = fmaxf(sqrtf(na2_sh), 1e-6f);
        float mx = -INFINITY;
        for (int s = lane; s < S; s += 32) {
            float ns  = fmaxf(sqrtf(ns2v[s]), 1e-6f);
            float sim = dotv[s] / (na * ns);
            dotv[s] = sim;
            mx = fmaxf(mx, sim);
        }
        for (int off = 16; off > 0; off >>= 1) mx = fmaxf(mx, __shfl_xor(mx, off, 32));
        float se = 0.f;
        for (int s = lane; s < S; s += 32) se += expf(dotv[s] - mx);
        for (int off = 16; off > 0; off >>= 1) se += __shfl_xor(se, off, 32);
        if (lane == 0) {
            float logp0 = dotv[0] - mx - logf(se);
            per_batch[b] = -logp0;
        }
    }
}

__global__ void __launch_bounds__(256)
reduce_mean_kernel(const float* __restrict__ per_batch, float* __restrict__ out, int B)
{
    __shared__ float red[256];
    float s = 0.f;
    for (int i = threadIdx.x; i < B; i += blockDim.x) s += per_batch[i];
    red[threadIdx.x] = s;
    __syncthreads();
    for (int st = 128; st > 0; st >>= 1) {
        if ((int)threadIdx.x < st) red[threadIdx.x] += red[threadIdx.x + st];
        __syncthreads();
    }
    if (threadIdx.x == 0) out[0] = red[0] / (float)B;
}

extern "C" void kernel_launch(void* const* d_in, const int* in_sizes, int n_in,
                              void* d_out, int out_size, void* d_ws, size_t ws_size,
                              hipStream_t stream)
{
    const float* anchor    = (const float*)d_in[0];
    const float* positives = (const float*)d_in[1];
    const float* negatives = (const float*)d_in[2];

    const int D = DFIX;                       // reference: D = 1024
    const int B = in_sizes[0] / D;            // anchor is (B,1,D)
    const int P = in_sizes[1] / (B * D);      // positives (B,P,D)
    const int N = in_sizes[2] / (B * D);      // negatives (B,N,D)
    const int S = N + 1;                      // 1 positive + N negatives
    const int TILES = (S + 15) / 16;          // 16-sample WMMA tiles per batch

    float* per_batch = (float*)d_ws;          // B floats of scratch

    dim3 grid(B), block(TILES * 32);          // one wave per tile (wave32)
    contrastive_wmma_kernel<<<grid, block, 0, stream>>>(
        anchor, positives, negatives, per_batch, P, S);
    reduce_mean_kernel<<<1, 256, 0, stream>>>(per_batch, (float*)d_out, B);
}
